// MaxViT3DMHSA_81887846466144
// MI455X (gfx1250) — compile-verified, matrix operand order checked by reading, not run on hardware
//
#include <hip/hip_runtime.h>

// ---------------------------------------------------------------------------
// CDNA5 (gfx1250) fused windowed-MHSA: v_wmma_f32_16x16x32_bf16 everywhere,
// TDM (tensor_load_to_lds) staging of V, wave32 layout throughout.
// ---------------------------------------------------------------------------

typedef __attribute__((ext_vector_type(16))) __bf16 v16bf;
typedef __attribute__((ext_vector_type(8)))  float  v8f;
typedef __attribute__((ext_vector_type(4)))  unsigned int v4u;
typedef __attribute__((ext_vector_type(8)))  int    v8i;
typedef __attribute__((ext_vector_type(4)))  int    v4i;

#define NTOK 343         // tokens per 7x7x7 window
#define NPAD 352         // 22 tiles of 16
#define DIM  256
#define HEADS 8
#define DH   32
#define NWIN 128
#define MROWS 43904      // 128*343

static __device__ __forceinline__ v8f wmma_bf16(v16bf a, v16bf b, v8f c) {
    return __builtin_amdgcn_wmma_f32_16x16x32_bf16(
        false, a, false, b, (short)0, c, false, false);
}

// 16x32 bf16 fragment from row-major bf16 (stride in elements).
// Lanes 0-15: row r, K {k0..k0+7, k0+16..k0+23}; lanes 16-31: K {+8, +24}.
template <typename PT>
static __device__ __forceinline__ v16bf load_frag_bf16(PT base, int stride,
                                                       int r, int k0, int lane) {
    PT p = base + (size_t)r * stride + k0 + ((lane >> 4) << 3);
    v16bf f;
#pragma unroll
    for (int i = 0; i < 8; ++i) { f[i] = p[i]; f[i + 8] = p[i + 16]; }
    return f;
}

// ---------------------------------------------------------------------------
// Kernel 0: transpose + convert weights to bf16.
// ---------------------------------------------------------------------------
__global__ void convert_weights(const float* __restrict__ w_qkv,
                                const float* __restrict__ w_out,
                                __bf16* __restrict__ wqkvT,
                                __bf16* __restrict__ woutT) {
    int idx = blockIdx.x * 256 + threadIdx.x;
    if (idx < 768 * 256) {
        int n = idx >> 8, k = idx & 255;
        wqkvT[idx] = (__bf16)w_qkv[k * 768 + n];
    }
    if (idx < 256 * 256) {
        int n = idx >> 8, k = idx & 255;
        woutT[idx] = (__bf16)w_out[k * 256 + n];
    }
}

// ---------------------------------------------------------------------------
// Kernel 1: QKV projection. X (43904x256 f32) @ Wqkv -> qkv bf16 scattered
// into [win][q|k|v][head][token][32]. grid(2744), block 256 (8 waves).
// Each wave: 16(M) x 96(N) via 6 accumulators, A-fragment reused 6x.
// ---------------------------------------------------------------------------
__global__ __launch_bounds__(256) void qkv_gemm(const float* __restrict__ x,
                                                const __bf16* __restrict__ wT,
                                                __bf16* __restrict__ qkv) {
    int lane = threadIdx.x & 31;
    int wave = threadIdx.x >> 5;
    int m0 = blockIdx.x * 16;
    int n0 = wave * 96;                 // 8 waves x 96 = 768 columns

    v8f acc[6];
#pragma unroll
    for (int t = 0; t < 6; ++t) acc[t] = (v8f){};

    const float* arow = x + (size_t)(m0 + (lane & 15)) * DIM + ((lane >> 4) << 3);
#pragma unroll
    for (int k0 = 0; k0 < DIM; k0 += 32) {
        if (k0 + 64 < DIM) __builtin_prefetch(arow + k0 + 64, 0, 3);
        v16bf a;
        {
            const float* p = arow + k0;
#pragma unroll
            for (int i = 0; i < 8; ++i) { a[i] = (__bf16)p[i]; a[i + 8] = (__bf16)p[i + 16]; }
        }
#pragma unroll
        for (int t = 0; t < 6; ++t) {
            v16bf b = load_frag_bf16(wT, DIM, n0 + t * 16 + (lane & 15), k0, lane);
            acc[t] = wmma_bf16(a, b, acc[t]);
        }
    }

    int nl = lane & 15, hs = lane >> 4;
#pragma unroll
    for (int v = 0; v < 8; ++v) {
        int row = m0 + v + 8 * hs;
        int win = row / NTOK, tok = row - win * NTOK;
#pragma unroll
        for (int t = 0; t < 6; ++t) {
            int col = n0 + t * 16 + nl;         // 0..767
            int sel  = col >> 8;                // 0:q 1:k 2:v
            int head = (col >> 5) & 7;
            int dhi  = col & 31;
            qkv[((((size_t)win * 3 + sel) * HEADS + head) * NTOK + tok) * DH + dhi] =
                (__bf16)acc[t][v];
        }
    }
}

// ---------------------------------------------------------------------------
// Kernel 2: attention per (q-block, head, window). block 128 (4 waves).
// V staged into LDS by the Tensor Data Mover, transposed LDS->LDS.
// ---------------------------------------------------------------------------
__global__ __launch_bounds__(128) void attn_kernel(const __bf16* __restrict__ qkv,
                                                   const float* __restrict__ bias_table,
                                                   __bf16* __restrict__ ctx) {
    __shared__ float  sS[64 * NPAD];      // logits f32          (90112 B)
    __shared__ __bf16 sP[64 * NPAD];      // probs bf16          (45056 B)
    __shared__ __bf16 sVT[DH * NPAD];     // V^T [dh][token]     (22528 B)
    __shared__ __bf16 sVraw[NTOK * DH];   // TDM landing buffer  (21952 B)
    __shared__ float  sInv[64];
    __shared__ float  sRed[128];
    __shared__ float  sSum[128];

    int qb   = blockIdx.x;   // 0..5
    int head = blockIdx.y;   // 0..7
    int win  = blockIdx.z;   // 0..127
    int tid  = threadIdx.x;
    int lane = tid & 31;
    int wave = tid >> 5;     // 0..3
    int q0   = qb * 64;

    const __bf16* qbase = qkv + (((size_t)win * 3 + 0) * HEADS + head) * NTOK * DH;
    const __bf16* kbase = qkv + (((size_t)win * 3 + 1) * HEADS + head) * NTOK * DH;
    const __bf16* vbase = qkv + (((size_t)win * 3 + 2) * HEADS + head) * NTOK * DH;

    // ---- phase 1: wave 0 kicks off a TDM load of V (343x32 bf16) into LDS ----
    if (wave == 0) {
        unsigned long long ga = (unsigned long long)(uintptr_t)vbase;
        unsigned int ldsoff = (unsigned int)(uintptr_t)(&sVraw[0]);  // addr[31:0] = LDS offset
        v4u g0;
        g0[0] = 1u;                                          // count = 1 valid descriptor
        g0[1] = ldsoff;                                      // lds_addr
        g0[2] = (unsigned int)(ga & 0xffffffffu);            // global_addr[31:0]
        g0[3] = (unsigned int)((ga >> 32) & 0x01ffffffu)     // global_addr[56:32]
                | 0x80000000u;                               // type = 2 ("image")
        v8i g1;
        g1[0] = 0x00010000;                                  // data_size = 2 bytes
        g1[1] = (int)(32u << 16);                            // tensor_dim0 = 32
        g1[2] = (int)((unsigned)NTOK << 16);                 // tensor_dim1 = 343
        g1[3] = (int)(32u << 16);                            // tile_dim0  = 32
        g1[4] = NTOK;                                        // tile_dim1  = 343
        g1[5] = 32;                                          // tensor_dim0_stride = 32
        g1[6] = 0;
        g1[7] = 0;
        v4i z4 = {};
        v8i z8 = {};
        __builtin_amdgcn_tensor_load_to_lds(g0, g1, z4, z4, z8, 0);
    }

    // ---- phase 2: S = q @ k^T * scale + bias -> LDS (overlaps the TDM) ----
    int qr0 = q0 + wave * 16;
    int nl = lane & 15, hs = lane >> 4;
    v16bf qa;
    {
        int r = qr0 + nl;
        if (r > NTOK - 1) r = NTOK - 1;       // rows >= NTOK discarded at store
        qa = load_frag_bf16(qbase, DH, r, 0, lane);
    }
    const float scale = 0.17677669529663687f; // 1/sqrt(32)
#pragma unroll 1
    for (int nt = 0; nt < NPAD / 16; ++nt) {
        int n0 = nt * 16;
        v16bf kb;
        {
            int r = n0 + nl;
            if (r > NTOK - 1) r = NTOK - 1;
            kb = load_frag_bf16(kbase, DH, r, 0, lane);
        }
        v8f acc = {};
        acc = wmma_bf16(qa, kb, acc);

        int j = n0 + nl;                      // key index
#pragma unroll
        for (int v = 0; v < 8; ++v) {
            int mloc = v + 8 * hs;
            int i = qr0 + mloc;               // query index
            float val;
            if (i >= NTOK) {
                val = 0.0f;                   // dead row, keep finite
            } else if (j >= NTOK) {
                val = -3.0e38f;               // masked key
            } else {
                int a1 = i / 49, ri = i - a1 * 49, a2 = ri / 7, a3 = ri - a2 * 7;
                int b1 = j / 49, rj = j - b1 * 49, b2 = rj / 7, b3 = rj - b2 * 7;
                int bidx = ((a1 - b1 + 6) * 13 + (a2 - b2 + 6)) * 13 + (a3 - b3 + 6);
                val = acc[v] * scale + bias_table[bidx * HEADS + head];
            }
            sS[(wave * 16 + mloc) * NPAD + j] = val;
        }
    }
    if (wave == 0) __builtin_amdgcn_s_wait_tensorcnt(0);
    __syncthreads();

    // ---- phase 3a: transpose V in LDS: sVraw[tok][dh] -> sVT[dh][tok] ----
    for (int idx = tid; idx < DH * NPAD; idx += 128) {
        int dhi = idx / NPAD;
        int tok = idx - dhi * NPAD;
        sVT[idx] = (tok < NTOK) ? sVraw[tok * DH + dhi] : (__bf16)0.0f;
    }

    // ---- phase 3b: row softmax, 2 threads per row ----
    {
        int r = tid >> 1, half = tid & 1;
        int c0 = half * 176;
        int c1 = c0 + 176; if (c1 > NTOK) c1 = NTOK;
        const float* row = &sS[r * NPAD];
        float m = -3.4e38f;
        for (int c = c0; c < c1; ++c) m = fmaxf(m, row[c]);
        sRed[tid] = m;
        __syncthreads();
        m = fmaxf(sRed[r * 2], sRed[r * 2 + 1]);
        float s = 0.0f;
        __bf16* prow = &sP[r * NPAD];
        for (int c = c0; c < c1; ++c) {
            float e = __expf(row[c] - m);
            s += e;
            prow[c] = (__bf16)e;
        }
        if (half == 1)
            for (int c = NTOK; c < NPAD; ++c) prow[c] = (__bf16)0.0f;
        sSum[tid] = s;
        __syncthreads();
        if (half == 0) {
            float tot = sSum[tid] + sSum[tid + 1];
            sInv[r] = (tot > 0.0f) ? (1.0f / tot) : 0.0f;
        }
    }
    __syncthreads();

    // ---- phase 4: ctx = P @ V  (16 x 32 per wave) ----
    v8f c0 = {}, c1 = {};
    int pr0 = wave * 16;
#pragma unroll 1
    for (int kt = 0; kt < NPAD / 32; ++kt) {
        int k0 = kt * 32;
        v16bf pa  = load_frag_bf16((const __bf16*)sP,  NPAD, pr0 + nl, k0, lane);
        v16bf vb0 = load_frag_bf16((const __bf16*)sVT, NPAD, nl,      k0, lane);
        v16bf vb1 = load_frag_bf16((const __bf16*)sVT, NPAD, nl + 16, k0, lane);
        c0 = wmma_bf16(pa, vb0, c0);
        c1 = wmma_bf16(pa, vb1, c1);
    }

#pragma unroll
    for (int v = 0; v < 8; ++v) {
        int mloc = v + 8 * hs;
        int i = qr0 + mloc;
        if (i < NTOK) {
            float inv = sInv[wave * 16 + mloc];
            size_t base = ((size_t)(win * NTOK + i)) * DIM + head * DH;
            ctx[base + nl]      = (__bf16)(c0[v] * inv);
            ctx[base + 16 + nl] = (__bf16)(c1[v] * inv);
        }
    }
}

// ---------------------------------------------------------------------------
// Kernel 3: output projection. ctx (43904x256 bf16) @ Wout -> out f32.
// grid(2744), block 256 (8 waves); each wave 16x32 via 2 accumulators.
// ---------------------------------------------------------------------------
__global__ __launch_bounds__(256) void out_gemm(const __bf16* __restrict__ ctx,
                                                const __bf16* __restrict__ wT,
                                                float* __restrict__ out) {
    int lane = threadIdx.x & 31;
    int wave = threadIdx.x >> 5;
    int m0 = blockIdx.x * 16;
    int n0 = wave * 32;                 // 8 waves x 32 = 256 columns

    v8f acc0 = {}, acc1 = {};
#pragma unroll
    for (int k0 = 0; k0 < DIM; k0 += 32) {
        v16bf a  = load_frag_bf16(ctx, DIM, m0 + (lane & 15), k0, lane);
        v16bf b0 = load_frag_bf16(wT,  DIM, n0 + (lane & 15),      k0, lane);
        v16bf b1 = load_frag_bf16(wT,  DIM, n0 + 16 + (lane & 15), k0, lane);
        acc0 = wmma_bf16(a, b0, acc0);
        acc1 = wmma_bf16(a, b1, acc1);
    }

    int nl = lane & 15, hs = lane >> 4;
#pragma unroll
    for (int v = 0; v < 8; ++v) {
        int row = m0 + v + 8 * hs;
        out[(size_t)row * DIM + n0 + nl]      = acc0[v];
        out[(size_t)row * DIM + n0 + 16 + nl] = acc1[v];
    }
}

// ---------------------------------------------------------------------------
// Host launcher. Inputs: x, w_qkv, w_out, rel_bias_table (all f32).
// ---------------------------------------------------------------------------
extern "C" void kernel_launch(void* const* d_in, const int* in_sizes, int n_in,
                              void* d_out, int out_size, void* d_ws, size_t ws_size,
                              hipStream_t stream) {
    const float* x          = (const float*)d_in[0];
    const float* w_qkv      = (const float*)d_in[1];
    const float* w_out      = (const float*)d_in[2];
    const float* bias_table = (const float*)d_in[3];

    char* ws = (char*)d_ws;
    __bf16* wqkvT = (__bf16*)ws;                 ws += (size_t)768 * 256 * 2;
    __bf16* woutT = (__bf16*)ws;                 ws += (size_t)256 * 256 * 2;
    __bf16* qkv   = (__bf16*)ws;                 ws += (size_t)NWIN * 3 * HEADS * NTOK * DH * 2;
    __bf16* ctxb  = (__bf16*)ws;                 ws += (size_t)MROWS * DIM * 2;

    convert_weights<<<768, 256, 0, stream>>>(w_qkv, w_out, wqkvT, woutT);
    qkv_gemm<<<dim3(MROWS / 16), 256, 0, stream>>>(x, wqkvT, qkv);
    attn_kernel<<<dim3(6, HEADS, NWIN), 128, 0, stream>>>(qkv, bias_table, ctxb);
    out_gemm<<<dim3(MROWS / 16), 256, 0, stream>>>(ctxb, woutT, (float*)d_out);
}
